// SwinUNETR_86749749445098
// MI455X (gfx1250) — matrix operand order repaired and checked by reading, not compiled, not run on hardware
//
#include <hip/hip_runtime.h>
#include <hip/hip_bf16.h>

typedef __attribute__((ext_vector_type(16))) _Float16 v16h;
typedef __attribute__((ext_vector_type(8)))  float    v8f;

#define SHIFT_ 3
#define NH_  3
#define C_   96
#define IMG_ 224
#define NWIN_ 8192            // 8 * 32 * 32
#define NTOK_ 401408          // 8*224*224
#define QKV_SCALE 0.17677669529663687f   // 32^-0.5

#ifndef __has_builtin
#define __has_builtin(x) 0
#endif
#if __has_builtin(__builtin_amdgcn_tensor_load_to_lds) && __has_builtin(__builtin_amdgcn_s_wait_tensorcnt)
#define HAVE_TDM 1
#else
#define HAVE_TDM 0
#endif

// ---------------- WMMA helpers (CDNA5 wave32 layouts) ----------------

__device__ __forceinline__ v8f wmma16(v16h a, v16h b, v8f c) {
    return __builtin_amdgcn_wmma_f32_16x16x32_f16(
        false, a, false, b, (short)0, c, false, false);
}

// A fragment 16x32 f16, row-major LDS, base = &A[m0*ld + k0]
__device__ __forceinline__ v16h load_a(const _Float16* __restrict__ base, int ld, int lane) {
    int row = lane & 15;
    int khi = (lane >> 4) << 3;     // 0 or 8
    v16h a;
#pragma unroll
    for (int v = 0; v < 8; ++v) {
        int kk = ((v < 4) ? (2 * v) : (16 + 2 * (v - 4))) + khi;
        a[2 * v]     = base[row * ld + kk];
        a[2 * v + 1] = base[row * ld + kk + 1];
    }
    return a;
}

// B fragment 32x16 f16 from transposed (N x K) LDS, base = &Bt[n0*ld + k0]
__device__ __forceinline__ v16h load_b(const _Float16* __restrict__ base, int ld, int lane) {
    int col = lane & 15;
    int kb  = (lane >> 4) << 3;     // 0 or 8
    v16h b;
#pragma unroll
    for (int v = 0; v < 8; ++v) {
        int kk = ((v < 4) ? (2 * v) : (16 + 2 * (v - 4))) + kb;
        b[2 * v]     = base[col * ld + kk];
        b[2 * v + 1] = base[col * ld + kk + 1];
    }
    return b;
}

// ---------------- TDM bulk LDS fill (wave-level DMA) ----------------

#if HAVE_TDM
typedef unsigned int u32x4 __attribute__((ext_vector_type(4)));
typedef int          i32x8 __attribute__((ext_vector_type(8)));
typedef int          i32x4 __attribute__((ext_vector_type(4)));

// 1-row 2D descriptor: data_size = 8B, tile_dim0 = n8 (< 65536) 8-byte units
__device__ __forceinline__ void tdm_load(const void* gsrc, unsigned lds_off, unsigned n8) {
    unsigned long long ga = (unsigned long long)(uintptr_t)gsrc;
    u32x4 g0;
    g0[0] = 1u;                                                 // count=1, user mode
    g0[1] = lds_off;                                            // lds_addr (bytes)
    g0[2] = (unsigned)ga;                                       // global_addr[31:0]
    g0[3] = ((unsigned)(ga >> 32) & 0x1FFFFFFu) | (2u << 30);   // addr[56:32] | type=2
    i32x8 g1;
    g1[0] = (int)(3u << 16);                                    // wg_mask=0, data_size=8B
    g1[1] = (int)((n8 & 0xFFFFu) << 16);                        // tensor_dim0 lo
    g1[2] = (int)((n8 >> 16) | (1u << 16));                     // tensor_dim0 hi | tensor_dim1=1
    g1[3] = (int)((n8 & 0xFFFFu) << 16);                        // tile_dim0 = n8
    g1[4] = 1;                                                  // tile_dim1=1, tile_dim2=0
    g1[5] = (int)n8;                                            // tensor_dim0_stride lo
    g1[6] = 0;                                                  // stride0 hi | stride1 lo
    g1[7] = 0;
    i32x4 z4 = {};
#if __clang_major__ >= 23
    i32x8 z8 = {};
    __builtin_amdgcn_tensor_load_to_lds(g0, g1, z4, z4, z8, 0);
#else
    __builtin_amdgcn_tensor_load_to_lds(g0, g1, z4, z4, 0);
#endif
}
#endif

// ---------------- weight prep: fp32 -> f16, transposed to (N x K) ----------------

__global__ __launch_bounds__(256) void prep_weights(
    const float* __restrict__ qkv_w,  // (96,288)
    const float* __restrict__ proj_w, // (96,96)
    const float* __restrict__ fc1_w,  // (96,384)
    const float* __restrict__ fc2_w,  // (384,96)
    _Float16* __restrict__ qkvWT,     // (288,96)
    _Float16* __restrict__ projWT,    // (96,96)
    _Float16* __restrict__ fc1WT,     // (384,96)
    _Float16* __restrict__ fc2WT)     // (96,384)
{
    int stride = gridDim.x * blockDim.x;
    int tid = blockIdx.x * blockDim.x + threadIdx.x;
    for (int i = tid; i < 288 * 96; i += stride) {
        int n = i / 96, k = i % 96;
        qkvWT[i] = (_Float16)qkv_w[k * 288 + n];
    }
    for (int i = tid; i < 96 * 96; i += stride) {
        int n = i / 96, k = i % 96;
        projWT[i] = (_Float16)proj_w[k * 96 + n];
    }
    for (int i = tid; i < 384 * 96; i += stride) {
        int n = i / 96, k = i % 96;
        fc1WT[i] = (_Float16)fc1_w[k * 384 + n];
    }
    for (int i = tid; i < 96 * 384; i += stride) {
        int n = i / 384, k = i % 384;
        fc2WT[i] = (_Float16)fc2_w[k * 96 + n];
    }
}

// ---------------- fused window attention: 2 windows per block ----------------
// rows 0..63 = window A, 64..127 = window B, stacked along M
// dynamic LDS layout (element offsets in _Float16 unless noted):
#define AH_X   0            // 128x96
#define AH_WQ  12288        // 288x96 qkv weights (NxK)
#define AH_WP  39936        // 96x96 proj weights (NxK)
#define AH_Q   49152        // 128x96
#define AH_K   61440        // 128x96
#define AH_VT  73728        // 2 x 96x64 (V transposed per window: dim x token)
#define AH_P   86016        // 128x64 softmax probs
#define AH_O   94208        // 128x96 attn output
#define A_SOFF 212992       // byte offset: f32 scores 128x64
#define A_ROFF 245760       // byte offset: f32 rpb stage 169*3
#define ATTN_LDS_BYTES 247808

__global__ __launch_bounds__(256) void attn_kernel(
    const float* __restrict__ x,        // (8,224,224,96)
    const float* __restrict__ qkv_b,    // (288)
    const float* __restrict__ proj_b,   // (96)
    const float* __restrict__ rpb,      // (169,3)
    const _Float16* __restrict__ qkvWT, // (288,96)
    const _Float16* __restrict__ projWT,// (96,96)
    float* __restrict__ xr)             // (8,224*224,96)
{
    extern __shared__ char smem[];
    _Float16* sh  = (_Float16*)smem;
    float*    sS  = (float*)(smem + A_SOFF);
    float*    sR  = (float*)(smem + A_ROFF);

    const int tid  = threadIdx.x;
    const int lane = tid & 31;
    const int wv   = tid >> 5;
    const int win0 = blockIdx.x * 2;

#if HAVE_TDM
    if (wv == 0) {
        tdm_load(qkvWT,  (unsigned)(uintptr_t)(sh + AH_WQ), (288 * 96 * 2) / 8);
        tdm_load(projWT, (unsigned)(uintptr_t)(sh + AH_WP), (96 * 96 * 2) / 8);
    }
#else
    for (int i = tid; i < 288 * 96; i += 256) sh[AH_WQ + i] = qkvWT[i];
    for (int i = tid; i < 96 * 96;  i += 256) sh[AH_WP + i] = projWT[i];
#endif

    // ---- stage 2 shifted windows (rows >=49 within each window zero-padded) ----
    for (int i = tid; i < 128 * 96; i += 256) {
        int R = i / 96, col = i - R * 96;
        int wloc = R >> 6, row = R & 63;
        _Float16 val = (_Float16)0.0f;
        if (row < 49) {
            int wa = win0 + wloc;
            int b  = wa >> 10, wh = (wa >> 5) & 31, ww = wa & 31;
            int r  = row / 7, cc = row - r * 7;
            int h0 = (wh * 7 + r  + SHIFT_) % IMG_;
            int w0 = (ww * 7 + cc + SHIFT_) % IMG_;
            val = (_Float16)x[(((size_t)b * IMG_ + h0) * IMG_ + w0) * C_ + col];
        }
        sh[AH_X + i] = val;
    }
    for (int i = tid; i < 169 * 3; i += 256) sR[i] = rpb[i];

#if HAVE_TDM
    if (wv == 0) __builtin_amdgcn_s_wait_tensorcnt(0);
#endif
    __syncthreads();

    // ---- QKV GEMM: (128x96) @ (96x288) ----
    for (int t = wv; t < 8 * 18; t += 8) {
        int mt = t / 18, nt = t - mt * 18;
        v8f acc = {};
#pragma unroll
        for (int k = 0; k < 3; ++k) {
            v16h a = load_a(sh + AH_X  + (mt * 16) * 96 + k * 32, 96, lane);
            v16h w = load_b(sh + AH_WQ + (nt * 16) * 96 + k * 32, 96, lane);
            acc = wmma16(a, w, acc);
        }
        int col   = nt * 16 + (lane & 15);
        int rbase = mt * 16 + ((lane & 16) ? 8 : 0);
        float bias = qkv_b[col];
#pragma unroll
        for (int r = 0; r < 8; ++r) {
            int R = rbase + r;
            float val = acc[r] + bias;
            if (col < 96)
                sh[AH_Q + R * 96 + col] = (_Float16)(val * QKV_SCALE);
            else if (col < 192)
                sh[AH_K + R * 96 + (col - 96)] = (_Float16)val;
            else {
                int wloc = R >> 6, token = R & 63;
                sh[AH_VT + wloc * 6144 + (col - 192) * 64 + token] = (_Float16)val;
            }
        }
    }
    __syncthreads();

    // ---- per-head attention (both windows) ----
    for (int h = 0; h < NH_; ++h) {
        // scores: Q(64x32) @ K^T(32x64) per window -> sS f32
        for (int t = wv; t < 32; t += 8) {
            int wloc = t >> 4, mt = (t >> 2) & 3, nt = t & 3;
            v16h a  = load_a(sh + AH_Q + (wloc * 64 + mt * 16) * 96 + 32 * h, 96, lane);
            v16h kf = load_b(sh + AH_K + (wloc * 64 + nt * 16) * 96 + 32 * h, 96, lane);
            v8f acc = {};
            acc = wmma16(a, kf, acc);
            int col   = nt * 16 + (lane & 15);
            int rbase = wloc * 64 + mt * 16 + ((lane & 16) ? 8 : 0);
#pragma unroll
            for (int r = 0; r < 8; ++r)
                sS[(rbase + r) * 64 + col] = acc[r];
        }
        __syncthreads();

        // softmax + rel-pos bias, one stacked row per thread (threads 0..127)
        if (tid < 128) {
            int R = tid, row = R & 63;
            if (row < 49) {
                int rh = row / 7, rw = row - rh * 7;
                float mx = -1e30f;
                for (int c = 0; c < 49; ++c) {
                    int ch = c / 7, cw = c - ch * 7;
                    int idx = (rh - ch + 6) * 13 + (rw - cw + 6);
                    float s = sS[R * 64 + c] + sR[idx * 3 + h];
                    mx = fmaxf(mx, s);
                }
                float sum = 0.0f;
                for (int c = 0; c < 49; ++c) {
                    int ch = c / 7, cw = c - ch * 7;
                    int idx = (rh - ch + 6) * 13 + (rw - cw + 6);
                    float e = __expf(sS[R * 64 + c] + sR[idx * 3 + h] - mx);
                    sum += e;
                    sS[R * 64 + c] = e;
                }
                float inv = 1.0f / sum;
                for (int c = 0; c < 64; ++c)
                    sh[AH_P + R * 64 + c] =
                        (c < 49) ? (_Float16)(sS[R * 64 + c] * inv) : (_Float16)0.0f;
            } else {
                for (int c = 0; c < 64; ++c)
                    sh[AH_P + R * 64 + c] = (_Float16)0.0f;
            }
        }
        __syncthreads();

        // O = P(64x64) @ V(64x32) per window -> sO columns [32h, 32h+32)
        for (int t = wv; t < 16; t += 8) {
            int wloc = t >> 3, mt = (t >> 1) & 3, nt = t & 1;
            v8f acc = {};
#pragma unroll
            for (int k = 0; k < 2; ++k) {
                v16h a  = load_a(sh + AH_P  + (wloc * 64 + mt * 16) * 64 + k * 32, 64, lane);
                v16h vb = load_b(sh + AH_VT + wloc * 6144 + (32 * h + nt * 16) * 64 + k * 32, 64, lane);
                acc = wmma16(a, vb, acc);
            }
            int col   = 32 * h + nt * 16 + (lane & 15);
            int rbase = wloc * 64 + mt * 16 + ((lane & 16) ? 8 : 0);
#pragma unroll
            for (int r = 0; r < 8; ++r)
                sh[AH_O + (rbase + r) * 96 + col] = (_Float16)acc[r];
        }
        __syncthreads();
    }

    // ---- proj: (128x96) @ (96x96), scatter with reverse shift ----
    for (int t = wv; t < 8 * 6; t += 8) {
        int mt = t / 6, nt = t - mt * 6;
        v8f acc = {};
#pragma unroll
        for (int k = 0; k < 3; ++k) {
            v16h a = load_a(sh + AH_O  + (mt * 16) * 96 + k * 32, 96, lane);
            v16h w = load_b(sh + AH_WP + (nt * 16) * 96 + k * 32, 96, lane);
            acc = wmma16(a, w, acc);
        }
        int col   = nt * 16 + (lane & 15);
        int rbase = mt * 16 + ((lane & 16) ? 8 : 0);
        float pb = proj_b[col];
#pragma unroll
        for (int r = 0; r < 8; ++r) {
            int R = rbase + r;
            int wloc = R >> 6, token = R & 63;
            if (token < 49) {
                int wa = win0 + wloc;
                int b  = wa >> 10, wh = (wa >> 5) & 31, ww = wa & 31;
                int rr = token / 7, cc = token - rr * 7;
                int h0 = (wh * 7 + rr + SHIFT_) % IMG_;
                int w0 = (ww * 7 + cc + SHIFT_) % IMG_;
                xr[(((size_t)b * IMG_ + h0) * IMG_ + w0) * C_ + col] = acc[r] + pb;
            }
        }
    }
}

// ---------------- fused LayerNorm + MLP + residual: 128 tokens per block ----------------
#define MH_XN  0            // 128x96  normalized f16
#define MH_W1  12288        // 384x96 fc1 (NxK)
#define MH_H   49152        // 128x384 hidden f16
#define MH_W2  98304        // 96x384 fc2 (NxK)
#define MLP_LDS_BYTES 270336

__global__ __launch_bounds__(256) void mlp_kernel(
    const float* __restrict__ xr,
    const float* __restrict__ norm2_g,
    const float* __restrict__ norm2_b,
    const float* __restrict__ fc1_b,
    const float* __restrict__ fc2_b,
    const _Float16* __restrict__ fc1WT,
    const _Float16* __restrict__ fc2WT,
    float* __restrict__ out)
{
    extern __shared__ char smem[];
    _Float16* sh = (_Float16*)smem;

    const int tid  = threadIdx.x;
    const int lane = tid & 31;
    const int wv   = tid >> 5;
    const size_t tok0 = (size_t)blockIdx.x * 128;

#if HAVE_TDM
    if (wv == 0) {
        tdm_load(fc1WT, (unsigned)(uintptr_t)(sh + MH_W1), (384 * 96 * 2) / 8);
        tdm_load(fc2WT, (unsigned)(uintptr_t)(sh + MH_W2), (96 * 384 * 2) / 8);
    }
#else
    for (int i = tid; i < 384 * 96; i += 256) sh[MH_W1 + i] = fc1WT[i];
    for (int i = tid; i < 96 * 384; i += 256) sh[MH_W2 + i] = fc2WT[i];
#endif

    // LayerNorm: one row per thread (threads 0..127)
    if (tid < 128) {
        size_t base = (tok0 + tid) * C_;
        float s = 0.0f, ss = 0.0f;
        for (int c = 0; c < C_; ++c) {
            float v = xr[base + c];
            s += v; ss += v * v;
        }
        float mean = s * (1.0f / C_);
        float var  = ss * (1.0f / C_) - mean * mean;
        float inv  = rsqrtf(var + 1e-5f);
        for (int c = 0; c < C_; ++c)
            sh[MH_XN + tid * 96 + c] =
                (_Float16)((xr[base + c] - mean) * inv * norm2_g[c] + norm2_b[c]);
    }
#if HAVE_TDM
    if (wv == 0) __builtin_amdgcn_s_wait_tensorcnt(0);
#endif
    __syncthreads();

    // fc1 + GELU: (128x96) @ (96x384)
    for (int t = wv; t < 8 * 24; t += 8) {
        int mt = t / 24, nt = t - mt * 24;
        v8f acc = {};
#pragma unroll
        for (int k = 0; k < 3; ++k) {
            v16h a = load_a(sh + MH_XN + (mt * 16) * 96 + k * 32, 96, lane);
            v16h w = load_b(sh + MH_W1 + (nt * 16) * 96 + k * 32, 96, lane);
            acc = wmma16(a, w, acc);
        }
        int col   = nt * 16 + (lane & 15);
        int rbase = mt * 16 + ((lane & 16) ? 8 : 0);
        float bias = fc1_b[col];
#pragma unroll
        for (int r = 0; r < 8; ++r) {
            float v = acc[r] + bias;
            float g = 0.5f * v * (1.0f + erff(v * 0.7071067811865476f));
            sh[MH_H + (rbase + r) * 384 + col] = (_Float16)g;
        }
    }
    __syncthreads();

    // fc2 + residual: (128x384) @ (384x96)
    for (int t = wv; t < 8 * 6; t += 8) {
        int mt = t / 6, nt = t - mt * 6;
        v8f acc = {};
#pragma unroll
        for (int k = 0; k < 12; ++k) {
            v16h a = load_a(sh + MH_H  + (mt * 16) * 384 + k * 32, 384, lane);
            v16h w = load_b(sh + MH_W2 + (nt * 16) * 384 + k * 32, 384, lane);
            acc = wmma16(a, w, acc);
        }
        int col   = nt * 16 + (lane & 15);
        int rbase = mt * 16 + ((lane & 16) ? 8 : 0);
        float bias = fc2_b[col];
#pragma unroll
        for (int r = 0; r < 8; ++r) {
            size_t idx = (tok0 + rbase + r) * C_ + col;
            out[idx] = acc[r] + bias + xr[idx];
        }
    }
}

// ---------------- host launch ----------------

extern "C" void kernel_launch(void* const* d_in, const int* in_sizes, int n_in,
                              void* d_out, int out_size, void* d_ws, size_t ws_size,
                              hipStream_t stream) {
    (void)in_sizes; (void)n_in; (void)out_size; (void)ws_size;

    const float* x       = (const float*)d_in[0];
    const float* qkv_w   = (const float*)d_in[1];
    const float* qkv_b   = (const float*)d_in[2];
    const float* proj_w  = (const float*)d_in[3];
    const float* proj_b  = (const float*)d_in[4];
    const float* rpb     = (const float*)d_in[5];
    const float* norm2_g = (const float*)d_in[6];
    const float* norm2_b = (const float*)d_in[7];
    const float* fc1_w   = (const float*)d_in[8];
    const float* fc1_b   = (const float*)d_in[9];
    const float* fc2_w   = (const float*)d_in[10];
    const float* fc2_b   = (const float*)d_in[11];

    char* ws = (char*)d_ws;
    _Float16* qkvWT  = (_Float16*)(ws + 0);          // 288*96*2  = 55296
    _Float16* projWT = (_Float16*)(ws + 55296);      // 96*96*2   = 18432
    _Float16* fc1WT  = (_Float16*)(ws + 73728);      // 384*96*2  = 73728
    _Float16* fc2WT  = (_Float16*)(ws + 147456);     // 96*384*2  = 73728
    float*    xr     = (float*)   (ws + 221184);     // 401408*96*4

    hipFuncSetAttribute((const void*)attn_kernel,
                        hipFuncAttributeMaxDynamicSharedMemorySize, ATTN_LDS_BYTES);
    hipFuncSetAttribute((const void*)mlp_kernel,
                        hipFuncAttributeMaxDynamicSharedMemorySize, MLP_LDS_BYTES);

    prep_weights<<<128, 256, 0, stream>>>(qkv_w, proj_w, fc1_w, fc2_w,
                                          qkvWT, projWT, fc1WT, fc2WT);

    attn_kernel<<<NWIN_ / 2, 256, ATTN_LDS_BYTES, stream>>>(
        x, qkv_b, proj_b, rpb, qkvWT, projWT, xr);

    mlp_kernel<<<NTOK_ / 128, 256, MLP_LDS_BYTES, stream>>>(
        xr, norm2_g, norm2_b, fc1_b, fc2_b, fc1WT, fc2WT, (float*)d_out);
}
